// RNN_37228776521966
// MI455X (gfx1250) — matrix operand order repaired and checked by reading, not compiled, not run on hardware
//
#include <hip/hip_runtime.h>
#include <hip/hip_bf16.h>
#include <math.h>

// ---------------------------------------------------------------------------
// Types for CDNA5 WMMA (gfx1250, wave32)
// ---------------------------------------------------------------------------
typedef __attribute__((ext_vector_type(16))) __bf16          v16bf;
typedef __attribute__((ext_vector_type(8)))  float           v8f;
typedef __attribute__((ext_vector_type(8)))  unsigned short  v8us;
typedef __attribute__((ext_vector_type(16))) unsigned short  v16us;

#define S_ 64
#define B_ 64
#define E_ 1024
#define H_ 1024
#define V_ 32000
#define L_ 2

// round-to-nearest-even f32 -> bf16 (stored as ushort)
__device__ __forceinline__ unsigned short f2bf(float f) {
  union { float f; unsigned u; } v; v.f = f;
  unsigned r = v.u + 0x7FFFu + ((v.u >> 16) & 1u);
  return (unsigned short)(r >> 16);
}

// ---------------------------------------------------------------------------
// Fragment loaders (bf16 stored as ushort, row-major, ld = row stride elems)
//
// A (16x32, MxK): lane L holds row m=L&15; half kh=L>>4 selects K-runs
//   [kh*8 .. kh*8+7] and [16+kh*8 .. 16+kh*8+7]  (two 16B contiguous loads)
// B (32x16, KxN) from W[N,K] row-major (B[k,n] = W[n][k]): lane L holds
//   col n=L&15, K-runs [kh*8 .. kh*8+7] and [16+kh*8 .. 16+kh*8+7] (two 16B loads)
// ---------------------------------------------------------------------------
__device__ __forceinline__ v16bf load_a_frag(const unsigned short* A, int mBase,
                                             int k0, int ld, int lane) {
  const int m  = lane & 15;
  const int kh = lane >> 4;
  const unsigned short* p = A + (size_t)(mBase + m) * ld + k0 + kh * 8;
  v8us lo = *(const v8us*)p;
  v8us hi = *(const v8us*)(p + 16);
  v16us c;
#pragma unroll
  for (int i = 0; i < 8; ++i) { c[i] = lo[i]; c[i + 8] = hi[i]; }
  return __builtin_bit_cast(v16bf, c);
}

__device__ __forceinline__ v16bf load_b_frag(const unsigned short* W, int nBase,
                                             int k0, int ld, int lane) {
  const int n  = lane & 15;
  const int kh = lane >> 4;
  const unsigned short* p = W + (size_t)(nBase + n) * ld + k0 + kh * 8;
  v8us lo = *(const v8us*)p;
  v8us hi = *(const v8us*)(p + 16);
  v16us c;
#pragma unroll
  for (int i = 0; i < 8; ++i) { c[i] = lo[i]; c[i + 8] = hi[i]; }
  return __builtin_bit_cast(v16bf, c);
}

__device__ __forceinline__ v8f wmma_bf16(v16bf a, v16bf b, v8f c) {
  return __builtin_amdgcn_wmma_f32_16x16x32_bf16(
      /*neg_a=*/false, a, /*neg_b=*/false, b,
      /*c_mod=*/(short)0, c, /*reuse_a=*/false, /*reuse_b=*/false);
}

// ---------------------------------------------------------------------------
// f32 -> bf16 conversion (grid-stride); NT loads for one-shot streams
// ---------------------------------------------------------------------------
__global__ void cvt_f32_bf16(const float* __restrict__ src,
                             unsigned short* __restrict__ dst, long n) {
  long i = (long)blockIdx.x * blockDim.x + threadIdx.x;
  long st = (long)gridDim.x * blockDim.x;
  for (; i < n; i += st) dst[i] = f2bf(__builtin_nontemporal_load(src + i));
}

// ---------------------------------------------------------------------------
// Embedding gather: ids [S*B], emb [V,E] f32 -> out [S*B, E] bf16
// one block per (t,b)
// ---------------------------------------------------------------------------
__global__ void embed_gather(const int* __restrict__ ids,
                             const float* __restrict__ emb,
                             unsigned short* __restrict__ out) {
  const int tb = blockIdx.x;
  const long id = (long)ids[tb];
  const float* src = emb + id * (long)E_;
  unsigned short* dst = out + (long)tb * E_;
  for (int i = threadIdx.x; i < E_; i += blockDim.x) dst[i] = f2bf(src[i]);
}

// ---------------------------------------------------------------------------
// One RNN layer step:  Out[B,H] = tanh(X[B,K]*W[H,K]^T + Hp[B,H]*R[H,H]^T + b1 + b2)
// K = H = 1024, B = 64.  Block = 4 waves (128 thr); wave tile = 16x16.
// grid.x = (B/16) * (H/64) = 4*16 = 64 blocks.
// Two accumulator chains per matmul to break WMMA RAW dependency.
// ---------------------------------------------------------------------------
__global__ void __launch_bounds__(128)
rnn_layer_step(const unsigned short* __restrict__ X,
               const unsigned short* __restrict__ W,
               const unsigned short* __restrict__ Hp,
               const unsigned short* __restrict__ R,
               const float* __restrict__ b1,
               const float* __restrict__ b2,
               unsigned short* __restrict__ Out,
               float* __restrict__ OutF32) {  // may be null
  const int lane = threadIdx.x & 31;
  const int wave = threadIdx.x >> 5;
  const int mBase = (blockIdx.x >> 4) * 16;                // 0..48
  const int nBase = (blockIdx.x & 15) * 64 + wave * 16;    // 0..1008

  v8f acc0 = {}, acc1 = {};
#pragma unroll 4
  for (int k0 = 0; k0 < 1024; k0 += 64) {
    v16bf a0 = load_a_frag(X, mBase, k0,      1024, lane);
    v16bf b0 = load_b_frag(W, nBase, k0,      1024, lane);
    v16bf a1 = load_a_frag(X, mBase, k0 + 32, 1024, lane);
    v16bf b1f = load_b_frag(W, nBase, k0 + 32, 1024, lane);
    acc0 = wmma_bf16(a0, b0, acc0);
    acc1 = wmma_bf16(a1, b1f, acc1);
  }
#pragma unroll 4
  for (int k0 = 0; k0 < 1024; k0 += 64) {
    v16bf a0 = load_a_frag(Hp, mBase, k0,      1024, lane);
    v16bf b0 = load_b_frag(R,  nBase, k0,      1024, lane);
    v16bf a1 = load_a_frag(Hp, mBase, k0 + 32, 1024, lane);
    v16bf b1f = load_b_frag(R,  nBase, k0 + 32, 1024, lane);
    acc0 = wmma_bf16(a0, b0, acc0);
    acc1 = wmma_bf16(a1, b1f, acc1);
  }

  const int n  = lane & 15;
  const int mh = lane >> 4;
  const int ng = nBase + n;
  const float bias = b1[ng] + b2[ng];
#pragma unroll
  for (int r = 0; r < 8; ++r) {
    const int m = mBase + r + 8 * mh;
    const float v = tanhf(acc0[r] + acc1[r] + bias);
    Out[(size_t)m * H_ + ng] = f2bf(v);
    if (OutF32) OutF32[(size_t)m * H_ + ng] = v;
  }
}

// ---------------------------------------------------------------------------
// Output projection: logits[M=4096, V=32000] = X1[M,1024] * outW[V,1024]^T + ob
// Block = 4 waves arranged 2x2 over a 64x128 block tile.
// Wave tile 32x64: 2 M-subtiles x 4 N-subtiles = 8 accumulators; each B
// fragment feeds 2 WMMAs -> 12 b128 loads per 8 WMMAs (L2-bandwidth bound,
// out_W stays resident in the 192MB L2).
// grid = (V/128, M/64) = (250, 64).
// Logits stores are nontemporal so the 524MB output stream does not evict
// the 65MB bf16 out_W from L2.
// ---------------------------------------------------------------------------
__global__ void __launch_bounds__(128)
logits_gemm(const unsigned short* __restrict__ X1,
            const unsigned short* __restrict__ Wo,
            const float* __restrict__ ob,
            float* __restrict__ outLogits) {
  const int lane = threadIdx.x & 31;
  const int wave = threadIdx.x >> 5;
  const int mBase = blockIdx.y * 64 + (wave >> 1) * 32;   // 2 M-subtiles of 16
  const int nBase = blockIdx.x * 128 + (wave & 1) * 64;   // 4 N-subtiles of 16

  v8f acc[2][4] = {{{}, {}, {}, {}}, {{}, {}, {}, {}}};
#pragma unroll 2
  for (int k0 = 0; k0 < 1024; k0 += 32) {
    v16bf a0 = load_a_frag(X1, mBase,      k0, 1024, lane);
    v16bf a1 = load_a_frag(X1, mBase + 16, k0, 1024, lane);
#pragma unroll
    for (int j = 0; j < 4; ++j) {
      v16bf b = load_b_frag(Wo, nBase + j * 16, k0, 1024, lane);
      acc[0][j] = wmma_bf16(a0, b, acc[0][j]);
      acc[1][j] = wmma_bf16(a1, b, acc[1][j]);
    }
  }

  const int n  = lane & 15;
  const int mh = lane >> 4;
#pragma unroll
  for (int j = 0; j < 4; ++j) {
    const int ng = nBase + j * 16 + n;
    const float bias = ob[ng];
#pragma unroll
    for (int i = 0; i < 2; ++i) {
#pragma unroll
      for (int r = 0; r < 8; ++r) {
        const int m = mBase + i * 16 + r + 8 * mh;
        __builtin_nontemporal_store(acc[i][j][r] + bias,
                                    outLogits + (long)m * V_ + ng);
      }
    }
  }
}

// ---------------------------------------------------------------------------
// Host side
// ---------------------------------------------------------------------------
extern "C" void kernel_launch(void* const* d_in, const int* in_sizes, int n_in,
                              void* d_out, int out_size, void* d_ws, size_t ws_size,
                              hipStream_t stream) {
  const int*   inputs = (const int*)  d_in[0];   // [S,B]
  const float* hidden = (const float*)d_in[1];   // [L,B,H]
  const float* emb    = (const float*)d_in[2];   // [V,E]
  const float* fc0_W  = (const float*)d_in[3];   // [H,E]
  const float* fc0_b  = (const float*)d_in[4];   // [H]
  const float* fc1_W  = (const float*)d_in[5];   // [H,H]
  const float* fc1_b  = (const float*)d_in[6];   // [H]
  const float* rec_W  = (const float*)d_in[7];   // [L,H,H]
  const float* rec_b  = (const float*)d_in[8];   // [L,H]
  const float* out_W  = (const float*)d_in[9];   // [V,H]
  const float* out_b  = (const float*)d_in[10];  // [V]

  float* logits = (float*)d_out;                           // [S,B,V]
  float* h_final = logits + (size_t)S_ * B_ * V_;          // [L,B,H]

  // workspace layout (bf16 buffers as ushort)
  char* ws = (char*)d_ws;
  size_t off = 0;
  auto alloc = [&](size_t bytes) { char* p = ws + off; off += (bytes + 255) & ~(size_t)255; return p; };
  unsigned short* wFC0  = (unsigned short*)alloc((size_t)H_ * E_ * 2);
  unsigned short* wFC1  = (unsigned short*)alloc((size_t)H_ * H_ * 2);
  unsigned short* wREC  = (unsigned short*)alloc((size_t)L_ * H_ * H_ * 2);
  unsigned short* wOUT  = (unsigned short*)alloc((size_t)V_ * H_ * 2);
  unsigned short* Xemb  = (unsigned short*)alloc((size_t)S_ * B_ * E_ * 2);
  unsigned short* X1all = (unsigned short*)alloc((size_t)S_ * B_ * H_ * 2);
  unsigned short* X0a   = (unsigned short*)alloc((size_t)B_ * H_ * 2);
  unsigned short* X0b   = (unsigned short*)alloc((size_t)B_ * H_ * 2);
  unsigned short* H1ini = (unsigned short*)alloc((size_t)B_ * H_ * 2);
  (void)ws_size; (void)n_in; (void)in_sizes; (void)out_size;

  // 1) convert weights + initial hidden to bf16
  cvt_f32_bf16<<<2048, 256, 0, stream>>>(fc0_W, wFC0, (long)H_ * E_);
  cvt_f32_bf16<<<2048, 256, 0, stream>>>(fc1_W, wFC1, (long)H_ * H_);
  cvt_f32_bf16<<<2048, 256, 0, stream>>>(rec_W, wREC, (long)L_ * H_ * H_);
  cvt_f32_bf16<<<8192, 256, 0, stream>>>(out_W, wOUT, (long)V_ * H_);
  cvt_f32_bf16<<<256,  256, 0, stream>>>(hidden,            X0a,   (long)B_ * H_);
  cvt_f32_bf16<<<256,  256, 0, stream>>>(hidden + (size_t)B_ * H_, H1ini, (long)B_ * H_);

  // 2) embedding gather -> bf16
  embed_gather<<<S_ * B_, 256, 0, stream>>>(inputs, emb, Xemb);

  unsigned short* x0slot[2] = {X0a, X0b};
  const unsigned short* rec0 = wREC;
  const unsigned short* rec1 = wREC + (size_t)H_ * H_;
  const float* rb0 = rec_b;
  const float* rb1 = rec_b + H_;

  // 3) sequential recurrence (deferred output projection)
  for (int t = 0; t < S_; ++t) {
    const unsigned short* xin = Xemb + (size_t)t * B_ * E_;
    unsigned short* h0_in  = x0slot[t & 1];
    unsigned short* h0_out = x0slot[(t & 1) ^ 1];
    float* f32_0 = (t == S_ - 1) ? h_final : nullptr;
    rnn_layer_step<<<64, 128, 0, stream>>>(xin, wFC0, h0_in, rec0,
                                           fc0_b, rb0, h0_out, f32_0);

    const unsigned short* h1_in = (t == 0) ? H1ini : X1all + (size_t)(t - 1) * B_ * H_;
    unsigned short* h1_out = X1all + (size_t)t * B_ * H_;
    float* f32_1 = (t == S_ - 1) ? (h_final + (size_t)B_ * H_) : nullptr;
    rnn_layer_step<<<64, 128, 0, stream>>>(h0_out, wFC1, h1_in, rec1,
                                           fc1_b, rb1, h1_out, f32_1);
  }

  // 4) one big logits GEMM: [4096 x 1024] * [1024 x 32000]
  dim3 grid(V_ / 128, (S_ * B_) / 64);
  logits_gemm<<<grid, 128, 0, stream>>>(X1all, wOUT, out_b, logits);
}
